// GraphNet_24739011625685
// MI455X (gfx1250) — matrix operand order repaired and checked by reading, not compiled, not run on hardware
//
#include <hip/hip_runtime.h>
#include <hip/hip_bf16.h>
#include <math.h>

#define N_NODES 8192
#define NF 32
#define KSPLIT 4
#define KCHUNK (N_NODES / KSPLIT)        // 2048 columns per wave
#define PACKW  (N_NODES / 16)            // packed dwords per adj row (512)

typedef _Float16 f16;
typedef __attribute__((ext_vector_type(8)))  _Float16 v8h;
typedef __attribute__((ext_vector_type(16))) _Float16 v16h;
typedef __attribute__((ext_vector_type(8)))  float    v8f;

static __device__ __forceinline__ v16h splat16(_Float16 v) {
    v16h r;
    #pragma unroll
    for (int i = 0; i < 16; ++i) r[i] = v;
    return r;
}

// ---------------------------------------------------------------------------
// Kernel 0: pack adj (int32, values 0..3) into 2-bit fields: 256MB -> 16MB.
// One thread produces one packed dword from 16 consecutive adj ints.
// ---------------------------------------------------------------------------
__global__ __launch_bounds__(256)
void pack_kernel(const int* __restrict__ adj, unsigned int* __restrict__ packed)
{
    const size_t idx = (size_t)blockIdx.x * 256 + threadIdx.x;   // dword index
    const int4* p = (const int4*)(adj + idx * 16);
    int4 q0 = p[0], q1 = p[1], q2 = p[2], q3 = p[3];
    int a[16] = { q0.x, q0.y, q0.z, q0.w,  q1.x, q1.y, q1.z, q1.w,
                  q2.x, q2.y, q2.z, q2.w,  q3.x, q3.y, q3.z, q3.w };
    unsigned int w = 0;
    #pragma unroll
    for (int t = 0; t < 16; ++t) w |= ((unsigned int)a[t] & 3u) << (2 * t);
    packed[idx] = w;
}

// ---------------------------------------------------------------------------
// Kernel 1: feature transform with thermometer re-basing, transposed output:
//   G1 = (X@w1)^T,  G2 = (X@w2 - 2 X@w1)^T,  G3 = (X@w3 - X@w2 - X@w1)^T
// so that  x*G1 + clamp01(x-1)*G2 + max(x-2,0)*G3  ==  mask_k selection.
// ---------------------------------------------------------------------------
__global__ __launch_bounds__(256)
void ht_kernel(const float* __restrict__ Vrow,     // [N,32] or nullptr
               const f16*   __restrict__ ZTin,     // [32,N] or nullptr
               const float* __restrict__ w1,
               const float* __restrict__ w2,
               const float* __restrict__ w3,
               f16* __restrict__ G1, f16* __restrict__ G2, f16* __restrict__ G3)
{
    const int j = blockIdx.x * blockDim.x + threadIdx.x;   // node index

    float x[NF];
    if (Vrow) {
        #pragma unroll
        for (int f = 0; f < NF; ++f) x[f] = Vrow[(size_t)j * NF + f];
    } else {
        #pragma unroll
        for (int f = 0; f < NF; ++f) x[f] = (float)ZTin[(size_t)f * N_NODES + j];
    }

    #pragma unroll 4
    for (int fo = 0; fo < NF; ++fo) {
        float a1 = 0.f, a2 = 0.f, a3 = 0.f;
        #pragma unroll
        for (int f = 0; f < NF; ++f) {
            a1 = fmaf(x[f], w1[f * NF + fo], a1);
            a2 = fmaf(x[f], w2[f * NF + fo], a2);
            a3 = fmaf(x[f], w3[f * NF + fo], a3);
        }
        const size_t o = (size_t)fo * N_NODES + j;
        G1[o] = (f16)a1;
        G2[o] = (f16)(a2 - 2.f * a1);
        G3[o] = (f16)(a3 - a2 - a1);
    }
}

// ---------------------------------------------------------------------------
// Kernel 2: masked aggregation via WMMA, K-split over grid.y.
//   O^T(partial) = sum_{j in slice} G_k (16x32 A) x maskfrag_k^T (32x16 B)
// One wave owns a 16-node tile; one packed b32 load yields the whole
// 16-column mask chunk for a lane; masks built with packed f16 min/max.
// Writes f32 partials; finalize kernel applies bias+relu+f16 cvt.
// ---------------------------------------------------------------------------
__global__ __launch_bounds__(64)
void agg_kernel(const unsigned int* __restrict__ packed,   // [N, 512] 2-bit adj
                const f16* __restrict__ G1,
                const f16* __restrict__ G2,
                const f16* __restrict__ G3,
                float* __restrict__ part)                  // [KSPLIT][32][N]
{
    const int lane      = threadIdx.x & 31;
    const int wave      = threadIdx.x >> 5;
    const int i0        = (blockIdx.x * (blockDim.x >> 5) + wave) * 16;
    const int rowInTile = lane & 15;
    const int hiHalf    = lane >> 4;                       // 0 or 1
    const size_t rowBase = (size_t)(i0 + rowInTile) * PACKW;
    const int jBase = blockIdx.y * KCHUNK;
    const int jEnd  = jBase + KCHUNK;

    v8f c0 = {}; v8f c1 = {};
    const f16* G[3] = { G1, G2, G3 };

    const v16h zero16 = splat16((f16)0.f);
    const v16h one16  = splat16((f16)1.f);
    const v16h two16  = splat16((f16)2.f);

    for (int j = jBase; j < jEnd; j += 32) {
        // one dword = this lane's 16 mask columns (2-bit fields)
        const unsigned int w = packed[rowBase + (j >> 4) + hiHalf];
        if (j + 512 < jEnd)
            __builtin_prefetch(&packed[rowBase + ((j + 512) >> 4) + hiHalf], 0, 1);

        // xv[t] = adj value in {0,1,2,3} as exact f16
        v16h xv;
        #pragma unroll
        for (int t = 0; t < 16; ++t)
            xv[t] = (f16)((w >> (2 * t)) & 3u);

        // thermometer masks (packed f16 min/max -> v_pk_{min,max}_num_f16)
        v16h m2 = __builtin_elementwise_min(
                      __builtin_elementwise_max(xv - one16, zero16), one16);
        v16h m3 = __builtin_elementwise_max(xv - two16, zero16);

        #pragma unroll
        for (int h = 0; h < 2; ++h) {
            #pragma unroll
            for (int k = 0; k < 3; ++k) {
                // A fragment: lane = feature row; two contiguous 16B loads cover
                // K {0..15} (lanes<16) / K {16..31} (lanes>=16).
                const f16* base = G[k] + (size_t)(h * 16 + rowInTile) * N_NODES
                                       + j + hiHalf * 16;
                v8h lo = *(const v8h*)base;
                v8h hi = *(const v8h*)(base + 8);
                v16h a;
                #pragma unroll
                for (int t = 0; t < 8; ++t) { a[t] = lo[t]; a[8 + t] = hi[t]; }

                v16h b = (k == 0) ? xv : ((k == 1) ? m2 : m3);
                if (h == 0)
                    c0 = __builtin_amdgcn_wmma_f32_16x16x32_f16(
                            false, a, false, b, (short)0, c0, false, false);
                else
                    c1 = __builtin_amdgcn_wmma_f32_16x16x32_f16(
                            false, a, false, b, (short)0, c1, false, false);
            }
        }
    }

    // store f32 partial O^T tile. D layout: VGPR r: lanes<16 -> M=r, lanes>=16 -> M=8+r.
    float* mypart = part + (size_t)blockIdx.y * NF * N_NODES;
    const int col = i0 + rowInTile;
    #pragma unroll
    for (int h = 0; h < 2; ++h) {
        v8f c = h ? c1 : c0;
        #pragma unroll
        for (int r = 0; r < 8; ++r) {
            const int feat = h * 16 + r + hiHalf * 8;
            mypart[(size_t)feat * N_NODES + col] = c[r];
        }
    }
}

// ---------------------------------------------------------------------------
// Kernel 3: finalize: sum K-split partials + bias, relu, store Z^T f16.
// ---------------------------------------------------------------------------
__global__ __launch_bounds__(256)
void fin_kernel(const float* __restrict__ part,   // [KSPLIT][32][N]
                const float* __restrict__ bias,   // [32]
                f16* __restrict__ ZTout)          // [32,N]
{
    const size_t idx = (size_t)blockIdx.x * 256 + threadIdx.x;   // 0..32*N-1
    const int feat = (int)(idx >> 13);                           // idx / N
    float s = bias[feat];
    #pragma unroll
    for (int k = 0; k < KSPLIT; ++k)
        s += part[(size_t)k * NF * N_NODES + idx];
    s = s > 0.f ? s : 0.f;
    ZTout[idx] = (f16)s;
}

// ---------------------------------------------------------------------------
// Kernel 4: sum-pool over nodes, per feature.
// ---------------------------------------------------------------------------
__global__ __launch_bounds__(256)
void pool_kernel(const f16* __restrict__ ZT2, float* __restrict__ pooled)
{
    __shared__ float s[256];
    const int f = blockIdx.x;
    float acc = 0.f;
    for (int j = threadIdx.x; j < N_NODES; j += 256)
        acc += (float)ZT2[(size_t)f * N_NODES + j];
    s[threadIdx.x] = acc;
    __syncthreads();
    #pragma unroll
    for (int stride = 128; stride > 0; stride >>= 1) {
        if (threadIdx.x < stride) s[threadIdx.x] += s[threadIdx.x + stride];
        __syncthreads();
    }
    if (threadIdx.x == 0) pooled[f] = s[0];
}

// ---------------------------------------------------------------------------
// Kernel 5: FC head: relu(pooled @ W0^T + b0) -> dot W1 -> sigmoid.
// ---------------------------------------------------------------------------
__global__ __launch_bounds__(32)
void fc_kernel(const float* __restrict__ pooled,
               const float* __restrict__ fcW0, const float* __restrict__ fcb0,
               const float* __restrict__ fcW1, const float* __restrict__ fcb1,
               float* __restrict__ out)
{
    __shared__ float h[32];
    const int t = threadIdx.x;
    float acc = fcb0[t];
    #pragma unroll
    for (int f = 0; f < 32; ++f) acc = fmaf(pooled[f], fcW0[t * 32 + f], acc);
    h[t] = acc > 0.f ? acc : 0.f;
    __syncthreads();
    if (t == 0) {
        float s = fcb1[0];
        #pragma unroll
        for (int i = 0; i < 32; ++i) s = fmaf(h[i], fcW1[i], s);
        out[0] = 1.f / (1.f + expf(-s));
    }
}

// ---------------------------------------------------------------------------
extern "C" void kernel_launch(void* const* d_in, const int* in_sizes, int n_in,
                              void* d_out, int out_size, void* d_ws, size_t ws_size,
                              hipStream_t stream)
{
    const float* V     = (const float*)d_in[0];
    const int*   adj   = (const int*)  d_in[1];
    const float* w1_0  = (const float*)d_in[2];
    const float* w2_0  = (const float*)d_in[3];
    const float* w3_0  = (const float*)d_in[4];
    const float* gb_0  = (const float*)d_in[5];
    const float* w1_1  = (const float*)d_in[6];
    const float* w2_1  = (const float*)d_in[7];
    const float* w3_1  = (const float*)d_in[8];
    const float* gb_1  = (const float*)d_in[9];
    const float* fcW_0 = (const float*)d_in[10];
    const float* fcb_0 = (const float*)d_in[11];
    const float* fcW_1 = (const float*)d_in[12];
    const float* fcb_1 = (const float*)d_in[13];

    char* ws = (char*)d_ws;
    const size_t HT_BYTES   = (size_t)NF * N_NODES * sizeof(f16);     // 512 KB
    const size_t PART_BYTES = (size_t)NF * N_NODES * sizeof(float);   // 1 MB
    f16*   G1     = (f16*)(ws + 0 * HT_BYTES);
    f16*   G2     = (f16*)(ws + 1 * HT_BYTES);
    f16*   G3     = (f16*)(ws + 2 * HT_BYTES);
    f16*   ZT1    = (f16*)(ws + 3 * HT_BYTES);
    f16*   ZT2    = (f16*)(ws + 4 * HT_BYTES);
    float* pooled = (float*)(ws + 5 * HT_BYTES);                      // 128 B
    float* part   = (float*)(ws + 3 * 1024 * 1024);                   // 4 MB
    unsigned int* packed = (unsigned int*)(ws + 8ull * 1024 * 1024);  // 16 MB

    const dim3 packGrid((N_NODES / 16) * (N_NODES / 256)), blk256(256);
    const dim3 htGrid(N_NODES / 256);
    const dim3 aggGrid(N_NODES / 32, KSPLIT), aggBlk(64);  // 2 waves/block
    const dim3 finGrid(NF * N_NODES / 256);

    pack_kernel<<<packGrid, blk256, 0, stream>>>(adj, packed);

    // Layer 1
    ht_kernel<<<htGrid, blk256, 0, stream>>>(V, nullptr, w1_0, w2_0, w3_0, G1, G2, G3);
    agg_kernel<<<aggGrid, aggBlk, 0, stream>>>(packed, G1, G2, G3, part);
    fin_kernel<<<finGrid, blk256, 0, stream>>>(part, gb_0, ZT1);
    // Layer 2
    ht_kernel<<<htGrid, blk256, 0, stream>>>(nullptr, ZT1, w1_1, w2_1, w3_1, G1, G2, G3);
    agg_kernel<<<aggGrid, aggBlk, 0, stream>>>(packed, G1, G2, G3, part);
    fin_kernel<<<finGrid, blk256, 0, stream>>>(part, gb_1, ZT2);
    // Head
    pool_kernel<<<dim3(32), blk256, 0, stream>>>(ZT2, pooled);
    fc_kernel<<<dim3(1), dim3(32), 0, stream>>>(pooled, fcW_0, fcb_0, fcW_1, fcb_1,
                                                (float*)d_out);
}
